// MiniKPCNN_57097295233597
// MI455X (gfx1250) — matrix operand order repaired and checked by reading, not compiled, not run on hardware
//
#include <hip/hip_runtime.h>
#include <hip/hip_bf16.h>

// ---------------------------------------------------------------------------
// MiniKPCNN for MI455X (gfx1250): bf16 WMMA conv stack + fused tail.
// Activations NHWC bf16 (12ch padded to 16), weights pre-swizzled into the
// exact V_WMMA_F32_16X16X32_BF16 B-fragment layout. Per-image pipeline so the
// working set (~200MB) stays inside the 192MB L2.
// Each wave computes TWO 16-pixel M-tiles so every B-fragment feeds 2 WMMAs.
// ---------------------------------------------------------------------------

typedef __attribute__((ext_vector_type(16))) __bf16 v16bf;
typedef __attribute__((ext_vector_type(8)))  __bf16 v8bf;
typedef __attribute__((ext_vector_type(8)))  float  v8f;

#define WIMG 1024
#define HIMG 1024
#define HW   (WIMG * HIMG)
#define TW   128           // output columns per workgroup (4 waves x 32 px)

__device__ __forceinline__ __bf16 f2bf(float f) {
    unsigned u = __builtin_bit_cast(unsigned, f);
    unsigned r = (u + 0x7FFFu + ((u >> 16) & 1u)) >> 16;   // RNE
    return __builtin_bit_cast(__bf16, (unsigned short)r);
}
__device__ __forceinline__ int reflect_i(int p, int n) {
    if (p < 0)  p = -p;
    if (p >= n) p = 2 * n - 2 - p;
    return p;
}
__device__ __forceinline__ v16bf cat8(v8bf lo, v8bf hi) {
    v16bf a;
#pragma unroll
    for (int q = 0; q < 8; ++q) { a[q] = lo[q]; a[8 + q] = hi[q]; }
    return a;
}

// ---------------------------------------------------------------------------
// Weight prep: convert f32 weights into bf16 B-fragment layouts.
// wp1: [50 taps][32 oc][16 ic]   (tap 49 = zeros, ic 12..15 = zeros)
// wp2: [25 taps][32 oc][32 ic]
// wp3: [25 taps][16 oc][32 ic]   (oc 8..15 = zeros)
// ---------------------------------------------------------------------------
__global__ __launch_bounds__(256) void weight_prep(
    const float* __restrict__ w1, const float* __restrict__ w2,
    const float* __restrict__ w3,
    __bf16* __restrict__ wp1, __bf16* __restrict__ wp2, __bf16* __restrict__ wp3)
{
    int i = blockIdx.x * 256 + threadIdx.x;
    if (i < 50 * 32 * 16) {
        int ic = i & 15, oc = (i >> 4) & 31, tap = i >> 9;
        float v = 0.f;
        if (tap < 49 && ic < 12) v = w1[(oc * 12 + ic) * 49 + tap];
        wp1[i] = f2bf(v);
        return;
    }
    int j = i - 50 * 32 * 16;
    if (j >= 0 && j < 25 * 32 * 32) {
        int ic = j & 31, oc = (j >> 5) & 31, tap = j >> 10;
        wp2[j] = f2bf(w2[(oc * 32 + ic) * 25 + tap]);
        return;
    }
    int k = i - 50 * 32 * 16 - 25 * 32 * 32;
    if (k >= 0 && k < 25 * 16 * 32) {
        int ic = k & 31, oc = (k >> 5) & 15, tap = k >> 9;
        float v = (oc < 8) ? w3[(oc * 32 + ic) * 25 + tap] : 0.f;
        wp3[k] = f2bf(v);
    }
}

// ---------------------------------------------------------------------------
// Preprocess: demodulate color, build 16ch NHWC bf16 net input + f32 color buf.
// ---------------------------------------------------------------------------
__global__ __launch_bounds__(256) void preprocess(
    const float* __restrict__ in, __bf16* __restrict__ netin,
    float* __restrict__ colorb)
{
    int pix = blockIdx.x * 256 + threadIdx.x;
    float a0 = in[3 * HW + pix], a1 = in[4 * HW + pix], a2 = in[5 * HW + pix];
    if (a0 < 0.001f) a0 = 1.f;
    if (a1 < 0.001f) a1 = 1.f;
    if (a2 < 0.001f) a2 = 1.f;
    float c0 = in[0 * HW + pix] / a0;
    float c1 = in[1 * HW + pix] / a1;
    float c2 = in[2 * HW + pix] / a2;
    union { __bf16 h[16]; uint4 q[2]; } v;
    v.h[0] = f2bf(c0); v.h[1] = f2bf(c1); v.h[2] = f2bf(c2);
    v.h[3] = f2bf(a0); v.h[4] = f2bf(a1); v.h[5] = f2bf(a2);
#pragma unroll
    for (int c = 6; c < 12; ++c) v.h[c] = f2bf(in[c * HW + pix]);
    v.h[12] = f2bf(0.f); v.h[13] = f2bf(0.f); v.h[14] = f2bf(0.f); v.h[15] = f2bf(0.f);
    uint4* dst = (uint4*)(netin + (size_t)pix * 16);
    dst[0] = v.q[0]; dst[1] = v.q[1];
    colorb[pix] = c0; colorb[HW + pix] = c1; colorb[2 * HW + pix] = c2;
}

// ---------------------------------------------------------------------------
// conv1: 16ch(12 padded) -> 32ch, 7x7, reflect pad 3, ReLU.  K-step = 2 taps.
// Per wave: 2 M-tiles (32 px) x 2 N-tiles (32 oc) -> 4 WMMAs per tap pair.
// ---------------------------------------------------------------------------
__global__ __launch_bounds__(128) void conv1_wmma(
    const __bf16* __restrict__ netin, const __bf16* __restrict__ wp1,
    const float* __restrict__ bias, __bf16* __restrict__ h1)
{
    __shared__ __bf16 tile[7 * (TW + 6) * 16];            // 30.0 KB
    const int tid = threadIdx.x;
    const int w0  = blockIdx.x * TW;
    const int h   = blockIdx.y;
    for (int s = tid; s < 7 * (TW + 6); s += 128) {       // stage reflect halo
        int r = s / (TW + 6), c = s % (TW + 6);
        int sr = reflect_i(h - 3 + r, HIMG);
        int sc = reflect_i(w0 - 3 + c, WIMG);
        const uint4* src = (const uint4*)(netin + ((size_t)sr * WIMG + sc) * 16);
        uint4* dst = (uint4*)(&tile[s * 16]);
        dst[0] = src[0]; dst[1] = src[1];
    }
    __syncthreads();

    const int lane = tid & 31, wave = tid >> 5;
    const int m = lane & 15, half = lane >> 4;
    const int col0 = wave * 32 + m;          // M-tile 0
    const int col1 = col0 + 16;              // M-tile 1

    float b0s = bias[m], b1s = bias[16 + m];
    v8f acc00, acc01, acc10, acc11;
#pragma unroll
    for (int g = 0; g < 8; ++g) {
        acc00[g] = b0s; acc01[g] = b1s; acc10[g] = b0s; acc11[g] = b1s;
    }

    for (int p = 0; p < 25; ++p) {
        int t0 = 2 * p;
        int t1 = (2 * p + 1 < 49) ? 2 * p + 1 : 48;       // pair 24: dup coords, zero W
        int ky0 = t0 / 7, kx0 = t0 % 7, ky1 = t1 / 7, kx1 = t1 % 7;
        int o0 = (ky0 * (TW + 6) + kx0) * 16 + half * 8;
        int o1 = (ky1 * (TW + 6) + kx1) * 16 + half * 8;
        v16bf a0 = cat8(*(const v8bf*)(&tile[o0 + col0 * 16]),
                        *(const v8bf*)(&tile[o1 + col0 * 16]));
        v16bf a1 = cat8(*(const v8bf*)(&tile[o0 + col1 * 16]),
                        *(const v8bf*)(&tile[o1 + col1 * 16]));
        int tsel = 2 * p;
        v16bf bf0 = cat8(*(const v8bf*)(wp1 + (tsel * 32 + m) * 16 + half * 8),
                         *(const v8bf*)(wp1 + ((tsel + 1) * 32 + m) * 16 + half * 8));
        v16bf bf1 = cat8(*(const v8bf*)(wp1 + (tsel * 32 + 16 + m) * 16 + half * 8),
                         *(const v8bf*)(wp1 + ((tsel + 1) * 32 + 16 + m) * 16 + half * 8));
        acc00 = __builtin_amdgcn_wmma_f32_16x16x32_bf16(false, a0, false, bf0,
                                                        (short)0, acc00, false, false);
        acc01 = __builtin_amdgcn_wmma_f32_16x16x32_bf16(false, a0, false, bf1,
                                                        (short)0, acc01, false, false);
        acc10 = __builtin_amdgcn_wmma_f32_16x16x32_bf16(false, a1, false, bf0,
                                                        (short)0, acc10, false, false);
        acc11 = __builtin_amdgcn_wmma_f32_16x16x32_bf16(false, a1, false, bf1,
                                                        (short)0, acc11, false, false);
    }
    size_t rowoff = (size_t)h * WIMG;
    int pcol = w0 + wave * 32 + half * 8;
#pragma unroll
    for (int g = 0; g < 8; ++g) {
        __bf16* d0 = h1 + (rowoff + pcol + g) * 32;
        __bf16* d1 = h1 + (rowoff + pcol + 16 + g) * 32;
        float v00 = acc00[g] > 0.f ? acc00[g] : 0.f;
        float v01 = acc01[g] > 0.f ? acc01[g] : 0.f;
        float v10 = acc10[g] > 0.f ? acc10[g] : 0.f;
        float v11 = acc11[g] > 0.f ? acc11[g] : 0.f;
        d0[m] = f2bf(v00); d0[16 + m] = f2bf(v01);
        d1[m] = f2bf(v10); d1[16 + m] = f2bf(v11);
    }
}

// ---------------------------------------------------------------------------
// 5x5 conv, 32 in-ch, reflect pad 2.  MODE 0: ->32ch bf16 + ReLU (conv2).
// MODE 1: ->8ch f32 logits (conv3, oc padded to 16 with zero weights).
// Per wave: 2 M-tiles; MODE 0 also 2 N-tiles -> 4 WMMAs per tap.
// ---------------------------------------------------------------------------
template <int MODE>
__global__ __launch_bounds__(128) void conv5x5_wmma(
    const __bf16* __restrict__ src, const __bf16* __restrict__ wp,
    const float* __restrict__ bias, __bf16* __restrict__ dstbf,
    float* __restrict__ dstf32)
{
    __shared__ __bf16 tile[5 * (TW + 4) * 32];            // 42.2 KB
    const int tid = threadIdx.x;
    const int w0  = blockIdx.x * TW;
    const int h   = blockIdx.y;
    for (int s = tid; s < 5 * (TW + 4); s += 128) {
        int r = s / (TW + 4), c = s % (TW + 4);
        int sr = reflect_i(h - 2 + r, HIMG);
        int sc = reflect_i(w0 - 2 + c, WIMG);
        const uint4* g = (const uint4*)(src + ((size_t)sr * WIMG + sc) * 32);
        uint4* d = (uint4*)(&tile[s * 32]);
        d[0] = g[0]; d[1] = g[1]; d[2] = g[2]; d[3] = g[3];
    }
    __syncthreads();

    const int lane = tid & 31, wave = tid >> 5;
    const int m = lane & 15, half = lane >> 4;
    const int col0 = wave * 32 + m;
    const int col1 = col0 + 16;
    const int OC = (MODE == 0) ? 32 : 16;

    float b0s = (MODE == 0) ? bias[m] : ((m < 8) ? bias[m] : 0.f);
    float b1s = (MODE == 0) ? bias[16 + m] : 0.f;
    v8f acc00, acc01, acc10, acc11;
#pragma unroll
    for (int g = 0; g < 8; ++g) {
        acc00[g] = b0s; acc01[g] = b1s; acc10[g] = b0s; acc11[g] = b1s;
    }

    for (int tap = 0; tap < 25; ++tap) {
        int ky = tap / 5, kx = tap % 5;
        int ob = (ky * (TW + 4) + kx) * 32;
        int base0 = ob + col0 * 32;
        int base1 = ob + col1 * 32;
        v16bf a0 = cat8(*(const v8bf*)(&tile[base0 + half * 8]),        // K 0..7 / 8..15
                        *(const v8bf*)(&tile[base0 + 16 + half * 8]));  // K 16..23 / 24..31
        v16bf a1 = cat8(*(const v8bf*)(&tile[base1 + half * 8]),
                        *(const v8bf*)(&tile[base1 + 16 + half * 8]));
        v16bf bf0 = cat8(*(const v8bf*)(wp + (tap * OC + m) * 32 + half * 8),
                         *(const v8bf*)(wp + (tap * OC + m) * 32 + 16 + half * 8));
        acc00 = __builtin_amdgcn_wmma_f32_16x16x32_bf16(false, a0, false, bf0,
                                                        (short)0, acc00, false, false);
        acc10 = __builtin_amdgcn_wmma_f32_16x16x32_bf16(false, a1, false, bf0,
                                                        (short)0, acc10, false, false);
        if (MODE == 0) {
            v16bf bf1 = cat8(*(const v8bf*)(wp + (tap * OC + 16 + m) * 32 + half * 8),
                             *(const v8bf*)(wp + (tap * OC + 16 + m) * 32 + 16 + half * 8));
            acc01 = __builtin_amdgcn_wmma_f32_16x16x32_bf16(false, a0, false, bf1,
                                                            (short)0, acc01, false, false);
            acc11 = __builtin_amdgcn_wmma_f32_16x16x32_bf16(false, a1, false, bf1,
                                                            (short)0, acc11, false, false);
        }
    }
    size_t rowoff = (size_t)h * WIMG;
    int pcol = w0 + wave * 32 + half * 8;
    if (MODE == 0) {
#pragma unroll
        for (int g = 0; g < 8; ++g) {
            __bf16* d0 = dstbf + (rowoff + pcol + g) * 32;
            __bf16* d1 = dstbf + (rowoff + pcol + 16 + g) * 32;
            float v00 = acc00[g] > 0.f ? acc00[g] : 0.f;
            float v01 = acc01[g] > 0.f ? acc01[g] : 0.f;
            float v10 = acc10[g] > 0.f ? acc10[g] : 0.f;
            float v11 = acc11[g] > 0.f ? acc11[g] : 0.f;
            d0[m] = f2bf(v00); d0[16 + m] = f2bf(v01);
            d1[m] = f2bf(v10); d1[16 + m] = f2bf(v11);
        }
    } else {
        if (m < 8) {
#pragma unroll
            for (int g = 0; g < 8; ++g) {
                dstf32[(rowoff + pcol + g) * 8 + m]      = acc00[g];
                dstf32[(rowoff + pcol + 16 + g) * 8 + m] = acc10[g];
            }
        }
    }
}

// ---------------------------------------------------------------------------
// Tail: softmax(logits/5) -> collapse 8 kernels against dwkernel -> depthwise
// 7x7 (zero pad) over demodulated color -> remodulate with albedo.
// ---------------------------------------------------------------------------
__global__ __launch_bounds__(256) void tail_fused(
    const float* __restrict__ logits, const float* __restrict__ colorb,
    const float* __restrict__ in, const float* __restrict__ dwk,
    float* __restrict__ out)
{
    __shared__ float sdwk[8 * 49];
    for (int s = threadIdx.x; s < 8 * 49; s += 256) sdwk[s] = dwk[s];
    __syncthreads();

    int pix = blockIdx.x * 256 + threadIdx.x;
    int row = pix >> 10, col = pix & (WIMG - 1);

    float z[8], mx = -3.4e38f;
#pragma unroll
    for (int k = 0; k < 8; ++k) { z[k] = logits[(size_t)pix * 8 + k] * 0.2f; mx = fmaxf(mx, z[k]); }
    float se = 0.f;
#pragma unroll
    for (int k = 0; k < 8; ++k) { z[k] = expf(z[k] - mx); se += z[k]; }
    float inv = 1.f / se;

    float wsum[49];
#pragma unroll
    for (int t = 0; t < 49; ++t) {
        float s = 0.f;
#pragma unroll
        for (int k = 0; k < 8; ++k) s += z[k] * sdwk[k * 49 + t];
        wsum[t] = s * inv;
    }
    float f0 = 0.f, f1 = 0.f, f2 = 0.f;
    for (int ky = 0; ky < 7; ++ky) {
        int rr = row + ky - 3;
        if ((unsigned)rr >= (unsigned)HIMG) continue;
        for (int kx = 0; kx < 7; ++kx) {
            int cc = col + kx - 3;
            if ((unsigned)cc >= (unsigned)WIMG) continue;
            float w = wsum[ky * 7 + kx];
            int q = rr * WIMG + cc;
            f0 += w * colorb[q];
            f1 += w * colorb[HW + q];
            f2 += w * colorb[2 * HW + q];
        }
    }
    float a0 = in[3 * HW + pix], a1 = in[4 * HW + pix], a2 = in[5 * HW + pix];
    if (a0 < 0.001f) a0 = 1.f;
    if (a1 < 0.001f) a1 = 1.f;
    if (a2 < 0.001f) a2 = 1.f;
    out[pix]          = a0 * f0;
    out[HW + pix]     = a1 * f1;
    out[2 * HW + pix] = a2 * f2;
}

// ---------------------------------------------------------------------------
// Host launcher: per-image pipeline, all on `stream` (graph-capture safe).
// ws layout (per-image buffers, reused across b):
//   netin  bf16 16ch : 32 MB @ 0
//   h1     bf16 32ch : 64 MB @ 32 MB
//   h2     bf16 32ch : 64 MB @ 96 MB
//   logits f32   8ch : 32 MB @ 160 MB
//   colorb f32   3ch : 12 MB @ 192 MB
//   wp1/wp2/wp3 bf16 : 128 KB @ 208 MB          (total ~208.2 MB)
// ---------------------------------------------------------------------------
extern "C" void kernel_launch(void* const* d_in, const int* in_sizes, int n_in,
                              void* d_out, int out_size, void* d_ws, size_t ws_size,
                              hipStream_t stream)
{
    (void)in_sizes; (void)n_in; (void)out_size; (void)ws_size;
    const float* input = (const float*)d_in[0];
    const float* w1 = (const float*)d_in[1];
    const float* b1 = (const float*)d_in[2];
    const float* w2 = (const float*)d_in[3];
    const float* b2 = (const float*)d_in[4];
    const float* w3 = (const float*)d_in[5];
    const float* b3 = (const float*)d_in[6];
    const float* dwk = (const float*)d_in[7];

    char* ws = (char*)d_ws;
    __bf16* netin  = (__bf16*)(ws);
    __bf16* h1     = (__bf16*)(ws + (size_t)32  * 1024 * 1024);
    __bf16* h2     = (__bf16*)(ws + (size_t)96  * 1024 * 1024);
    float*  logits = (float*)(ws + (size_t)160 * 1024 * 1024);
    float*  colorb = (float*)(ws + (size_t)192 * 1024 * 1024);
    __bf16* wp1    = (__bf16*)(ws + (size_t)208 * 1024 * 1024);
    __bf16* wp2    = wp1 + 50 * 32 * 16;
    __bf16* wp3    = wp2 + 25 * 32 * 32;

    weight_prep<<<250, 256, 0, stream>>>(w1, w2, w3, wp1, wp2, wp3);

    for (int b = 0; b < 4; ++b) {
        const float* inb = input + (size_t)b * 12 * HW;
        float* outb = (float*)d_out + (size_t)b * 3 * HW;

        preprocess<<<HW / 256, 256, 0, stream>>>(inb, netin, colorb);
        conv1_wmma<<<dim3(WIMG / TW, HIMG), 128, 0, stream>>>(netin, wp1, b1, h1);
        conv5x5_wmma<0><<<dim3(WIMG / TW, HIMG), 128, 0, stream>>>(h1, wp2, b2, h2, nullptr);
        conv5x5_wmma<1><<<dim3(WIMG / TW, HIMG), 128, 0, stream>>>(h2, wp3, b3, nullptr, logits);
        tail_fused<<<HW / 256, 256, 0, stream>>>(logits, colorb, inb, dwk, outb);
    }
}
